// SLPPolicyValueNet_7687991460579
// MI455X (gfx1250) — matrix operand order repaired and checked by reading, not compiled, not run on hardware
//
#include <hip/hip_runtime.h>
#include <hip/hip_bf16.h>
#include <math.h>

// ---------------------------------------------------------------------------
// Problem constants (match reference)
// ---------------------------------------------------------------------------
#define BB  32
#define NNN 512
#define DDD 256
#define HHH 8
#define HDD 32
#define MM  (BB * NNN)          // 16384 tokens
#define FF  (4 * DDD)           // 1024 ffn hidden

typedef __attribute__((ext_vector_type(16))) __bf16 v16bf;
typedef __attribute__((ext_vector_type(8)))  float  v8f;

__device__ __forceinline__ v8f wmma_bf16(v16bf a, v16bf b, v8f c) {
    // D = A(16x32 bf16) x B(32x16 bf16) + C(16x16 f32)
    return __builtin_amdgcn_wmma_f32_16x16x32_bf16(
        /*neg_a=*/false, a, /*neg_b=*/false, b,
        /*c_mod=*/(short)0, c, /*reuse_a=*/false, /*reuse_b=*/false);
}

__device__ __forceinline__ float gelu_f(float x) {
    return 0.5f * x * (1.0f + erff(x * 0.70710678118654752f));
}
__device__ __forceinline__ __bf16 to_bf(float x) { return (__bf16)x; }

// ---------------------------------------------------------------------------
// Generic WMMA GEMM:  C(M,N) = act(A(M,K) @ W(K,N) + bias [+ res])
// Block tile 64(M) x 256(N), K-step 32. 256 threads = 8 waves (wave32).
// Waves arranged 2(M) x 4(N); each wave computes a 32x64 tile
// (2 A-frags x 4 B-frags -> 8 WMMA 16x16x32 per K-step).
// Requires M%64==0, N%256==0, K%32==0 (true for all uses here).
// ---------------------------------------------------------------------------
template <int ACT, bool RES>
__global__ __launch_bounds__(256)
void gemm_wmma(const float* __restrict__ A, const float* __restrict__ W,
               const float* __restrict__ bias, const float* __restrict__ res,
               float* __restrict__ C, int M, int K, int N)
{
    __shared__ __bf16 sA[64 * 32];    // [row][k]
    __shared__ __bf16 sBT[256 * 32];  // [col][k]  (B transposed in LDS)

    const int t    = threadIdx.x;
    const int lane = t & 31;
    const int wave = t >> 5;       // 0..7
    const int wrow = wave >> 2;    // 0..1 -> 32 rows each
    const int wcol = wave & 3;     // 0..3 -> 64 cols each
    const int bm   = blockIdx.y * 64;
    const int bn   = blockIdx.x * 256;

    const int ln    = lane & 15;
    const int hh    = (lane & 16) >> 1;   // unused (0 or 8); A-frag k offset is khalf
    const int khalf = lane & 16;          // B-frag k offset (0 or 16)
    const int rhalf = (lane >> 4) & 1;    // C-frag row offset (0 or 8)

    v8f acc[2][4];
    #pragma unroll
    for (int r = 0; r < 2; ++r)
        #pragma unroll
        for (int c = 0; c < 4; ++c)
            #pragma unroll
            for (int g = 0; g < 8; ++g) acc[r][c][g] = 0.0f;

    for (int k0 = 0; k0 < K; k0 += 32) {
        // stage A tile (64x32) as bf16: 4 threads/row, 8 floats each
        {
            const int r  = t >> 2;
            const int c  = (t & 3) * 8;
            const float* src = A + (size_t)(bm + r) * K + k0 + c;
            __bf16* dst = sA + r * 32 + c;
            #pragma unroll
            for (int i = 0; i < 8; ++i) dst[i] = to_bf(src[i]);
        }
        // stage B tile (32x256) transposed: 8 threads/k-row, 32 cols each
        {
            const int kk = t >> 3;
            const int c0 = (t & 7) * 32;
            const float* src = W + (size_t)(k0 + kk) * N + bn + c0;
            #pragma unroll
            for (int i = 0; i < 32; ++i) sBT[(c0 + i) * 32 + kk] = to_bf(src[i]);
        }
        __syncthreads();

        v16bf a[2], b[4];
        #pragma unroll
        for (int r = 0; r < 2; ++r) {
            const __bf16* base = sA + (wrow * 32 + r * 16 + ln) * 32;
            #pragma unroll
            for (int i = 0; i < 16; ++i)
                a[r][i] = base[i + khalf];
        }
        #pragma unroll
        for (int c = 0; c < 4; ++c) {
            const __bf16* base = sBT + (wcol * 64 + c * 16 + ln) * 32 + khalf;
            #pragma unroll
            for (int i = 0; i < 16; ++i) b[c][i] = base[i];
        }
        #pragma unroll
        for (int r = 0; r < 2; ++r)
            #pragma unroll
            for (int c = 0; c < 4; ++c)
                acc[r][c] = wmma_bf16(a[r], b[c], acc[r][c]);
        __syncthreads();
    }

    // epilogue
    #pragma unroll
    for (int r = 0; r < 2; ++r) {
        #pragma unroll
        for (int c = 0; c < 4; ++c) {
            const int col = bn + wcol * 64 + c * 16 + ln;
            const float bcol = bias[col];
            #pragma unroll
            for (int g = 0; g < 8; ++g) {
                const int row = bm + wrow * 32 + r * 16 + g + rhalf * 8;
                float v = acc[r][c][g] + bcol;
                if (RES) v += res[(size_t)row * N + col];
                if (ACT == 1) v = gelu_f(v);
                C[(size_t)row * N + col] = v;
            }
        }
    }
}

// ---------------------------------------------------------------------------
// Fused masked attention. grid=(N/16, H, B), block = 1 wave (32 threads).
// Per block: 16 query rows. 32 WMMAs (QK^T) + 32 WMMAs (PV).
// ---------------------------------------------------------------------------
__global__ __launch_bounds__(32)
void attn_wmma(const float* __restrict__ Q, const float* __restrict__ K,
               const float* __restrict__ V, const float* __restrict__ adj,
               float* __restrict__ OUT)
{
    __shared__ float  sS[16 * NNN];   // masked scaled scores (32 KB)
    __shared__ __bf16 sP[16 * NNN];   // un-normalized probs   (16 KB)

    const int qt = blockIdx.x, h = blockIdx.y, b = blockIdx.z;
    const int lane  = threadIdx.x;
    const int ln    = lane & 15;
    const int hh    = (lane & 16) >> 1;
    const int khalf = lane & 16;
    const int rhalf = (lane >> 4) & 1;
    const float scale = 0.17677669529663687f;  // 1/sqrt(32)

    const size_t rowQ = (size_t)b * NNN + qt * 16;

    // A fragment of Q (16 rows x k=32), loaded once
    v16bf aq;
    {
        const float* qr = Q + (rowQ + ln) * DDD + h * HDD;
        #pragma unroll
        for (int i = 0; i < 16; ++i) aq[i] = to_bf(qr[i + khalf]);
    }

    float mx[8];
    #pragma unroll
    for (int g = 0; g < 8; ++g) mx[g] = -3.0e38f;

    // ---- pass 1: scores = scale * Q K^T, mask, store to LDS, track row max
    for (int kt = 0; kt < NNN / 16; ++kt) {
        v16bf bk;
        const float* kr = K + ((size_t)b * NNN + kt * 16 + ln) * DDD + h * HDD + khalf;
        #pragma unroll
        for (int i = 0; i < 16; ++i) bk[i] = to_bf(kr[i]);

        v8f s;
        #pragma unroll
        for (int g = 0; g < 8; ++g) s[g] = 0.0f;
        s = wmma_bf16(aq, bk, s);

        const int col = kt * 16 + ln;
        #pragma unroll
        for (int g = 0; g < 8; ++g) {
            const int r = g + rhalf * 8;
            float val = s[g] * scale;
            const float av = adj[((size_t)b * NNN + qt * 16 + r) * NNN + col];
            if (av == 0.0f) val = -__builtin_huge_valf();
            sS[r * NNN + col] = val;
            mx[g] = fmaxf(mx[g], val);
        }
    }
    // reduce row max across the 16 lanes of each half-wave
    #pragma unroll
    for (int g = 0; g < 8; ++g) {
        #pragma unroll
        for (int d = 1; d < 16; d <<= 1)
            mx[g] = fmaxf(mx[g], __shfl_xor(mx[g], d, 32));
    }
    __syncthreads();

    // ---- pass 2: exp, row sums, bf16 probs to LDS
    float sm[8];
    #pragma unroll
    for (int g = 0; g < 8; ++g) sm[g] = 0.0f;
    for (int kt = 0; kt < NNN / 16; ++kt) {
        const int col = kt * 16 + ln;
        #pragma unroll
        for (int g = 0; g < 8; ++g) {
            const int r = g + rhalf * 8;
            // val=-inf -> exp(-inf)=0 even when whole row is masked (mx finite)
            const float p = __expf(sS[r * NNN + col] - mx[g]);
            sm[g] += p;
            sP[r * NNN + col] = to_bf(p);
        }
    }
    #pragma unroll
    for (int g = 0; g < 8; ++g) {
        #pragma unroll
        for (int d = 1; d < 16; d <<= 1)
            sm[g] += __shfl_xor(sm[g], d, 32);
    }
    __syncthreads();

    // ---- P @ V   (16 x 512) @ (512 x 32), two 16-col output tiles
    v8f o0, o1;
    #pragma unroll
    for (int g = 0; g < 8; ++g) { o0[g] = 0.0f; o1[g] = 0.0f; }
    for (int kt = 0; kt < NNN / 32; ++kt) {
        v16bf ap;
        const __bf16* pr = sP + ln * NNN + kt * 32;
        #pragma unroll
        for (int i = 0; i < 16; ++i) ap[i] = pr[i + khalf];

        v16bf bv0, bv1;
        const float* vb = V + ((size_t)b * NNN + kt * 32 + khalf) * DDD + h * HDD;
        #pragma unroll
        for (int i = 0; i < 16; ++i) {
            const float* vr = vb + (size_t)i * DDD;
            bv0[i] = to_bf(vr[ln]);
            bv1[i] = to_bf(vr[16 + ln]);
        }
        o0 = wmma_bf16(ap, bv0, o0);
        o1 = wmma_bf16(ap, bv1, o1);
    }

    // ---- normalize and write (rows with zero sum -> 0, like nan_to_num)
    #pragma unroll
    for (int g = 0; g < 8; ++g) {
        const int r = g + rhalf * 8;
        const float inv = (sm[g] > 0.0f) ? (1.0f / sm[g]) : 0.0f;
        float* orow = OUT + (rowQ + r) * DDD + h * HDD;
        orow[ln]      = o0[g] * inv;
        orow[16 + ln] = o1[g] * inv;
    }
}

// ---------------------------------------------------------------------------
// Input projection (16 -> 256) + layernorm + gelu. grid = M rows, block 256.
// ---------------------------------------------------------------------------
__global__ __launch_bounds__(256)
void input_kernel(const float* __restrict__ feat, const float* __restrict__ w,
                  const float* __restrict__ b, const float* __restrict__ lng,
                  const float* __restrict__ lnb, float* __restrict__ out)
{
    __shared__ float f[16];
    __shared__ float r1[256], r2[256];
    const int row = blockIdx.x, t = threadIdx.x;
    if (t < 16) f[t] = feat[(size_t)row * 16 + t];
    __syncthreads();
    float acc = b[t];
    #pragma unroll
    for (int k = 0; k < 16; ++k) acc += f[k] * w[k * DDD + t];
    r1[t] = acc; r2[t] = acc * acc;
    __syncthreads();
    for (int s = 128; s > 0; s >>= 1) {
        if (t < s) { r1[t] += r1[t + s]; r2[t] += r2[t + s]; }
        __syncthreads();
    }
    const float mean = r1[0] * (1.0f / DDD);
    const float var  = r2[0] * (1.0f / DDD) - mean * mean;
    const float rstd = rsqrtf(var + 1e-5f);
    out[(size_t)row * DDD + t] = gelu_f((acc - mean) * rstd * lng[t] + lnb[t]);
}

// ---------------------------------------------------------------------------
// LayerNorm over last dim (256). grid = M rows, block 256.
// ---------------------------------------------------------------------------
__global__ __launch_bounds__(256)
void ln_kernel(const float* __restrict__ in, const float* __restrict__ g,
               const float* __restrict__ b, float* __restrict__ out)
{
    __shared__ float r1[256], r2[256];
    const int row = blockIdx.x, t = threadIdx.x;
    const float x = in[(size_t)row * DDD + t];
    r1[t] = x; r2[t] = x * x;
    __syncthreads();
    for (int s = 128; s > 0; s >>= 1) {
        if (t < s) { r1[t] += r1[t + s]; r2[t] += r2[t + s]; }
        __syncthreads();
    }
    const float mean = r1[0] * (1.0f / DDD);
    const float var  = r2[0] * (1.0f / DDD) - mean * mean;
    const float rstd = rsqrtf(var + 1e-5f);
    out[(size_t)row * DDD + t] = (x - mean) * rstd * g[t] + b[t];
}

// masked mean pooling: g[b,d] = sum_n h[b,n,d]*m[b,n] / (sum_n m[b,n] + 1e-8)
__global__ __launch_bounds__(256)
void pool_kernel(const float* __restrict__ h, const float* __restrict__ m,
                 float* __restrict__ g)
{
    const int b = blockIdx.x, t = threadIdx.x;
    float s = 0.0f, ms = 0.0f;
    for (int n = 0; n < NNN; ++n) {
        const float mv = m[(size_t)b * NNN + n];
        s  += h[((size_t)b * NNN + n) * DDD + t] * mv;
        ms += mv;
    }
    g[(size_t)b * DDD + t] = s / (ms + 1e-8f);
}

// small dense layer for (B x K) inputs. grid = B, block 256.
__global__ __launch_bounds__(256)
void small_linear(const float* __restrict__ A, const float* __restrict__ W,
                  const float* __restrict__ bias, float* __restrict__ C,
                  int Kdim, int Ndim, int act)
{
    const int b = blockIdx.x, j = threadIdx.x;
    if (j >= Ndim) return;
    float acc = bias[j];
    for (int k = 0; k < Kdim; ++k) acc += A[(size_t)b * Kdim + k] * W[(size_t)k * Ndim + j];
    if (act) acc = gelu_f(acc);
    C[(size_t)b * Ndim + j] = acc;
}

// value head: out[b] = dot(T1[b,:], w) + b0. 1 block, 32 threads.
__global__ __launch_bounds__(32)
void val2_kernel(const float* __restrict__ T1, const float* __restrict__ w,
                 const float* __restrict__ b0, float* __restrict__ out)
{
    const int b = threadIdx.x;
    float acc = b0[0];
    for (int k = 0; k < DDD; ++k) acc += T1[(size_t)b * DDD + k] * w[k];
    out[b] = acc;
}

// logits[b,n] = dot(qvec[b], kmat[b,n]) / 16, masked to -inf
__global__ __launch_bounds__(256)
void logits_kernel(const float* __restrict__ qvec, const float* __restrict__ kmat,
                   const float* __restrict__ mask, float* __restrict__ out)
{
    const int idx = blockIdx.x * 256 + threadIdx.x;  // b*N + n
    const int b = idx >> 9;
    const float* qp = qvec + (size_t)b * DDD;
    const float* kp = kmat + (size_t)idx * DDD;
    float acc = 0.0f;
    for (int d = 0; d < DDD; ++d) acc += qp[d] * kp[d];
    acc *= (1.0f / 16.0f);  // 1/sqrt(256)
    out[idx] = (mask[idx] == 0.0f) ? -__builtin_huge_valf() : acc;
}

// CAT[b] = concat(G[b], HF[b, u_idx[b]]). grid = B, block 256.
__global__ __launch_bounds__(256)
void cat_kernel(const float* __restrict__ G, const float* __restrict__ HF,
                const int* __restrict__ u_idx, float* __restrict__ CAT)
{
    const int b = blockIdx.x, t = threadIdx.x;
    CAT[(size_t)b * 512 + t] = G[(size_t)b * DDD + t];
    const int ui = u_idx[b];
    CAT[(size_t)b * 512 + 256 + t] = HF[((size_t)b * NNN + ui) * DDD + t];
}

// ---------------------------------------------------------------------------
// Host launcher. Params flattened in JAX pytree order (dict keys sorted).
// ---------------------------------------------------------------------------
extern "C" void kernel_launch(void* const* d_in, const int* in_sizes, int n_in,
                              void* d_out, int out_size, void* d_ws, size_t ws_size,
                              hipStream_t stream)
{
    const float* features   = (const float*)d_in[0];
    const float* adj        = (const float*)d_in[1];
    const float* valid_mask = (const float*)d_in[2];
    const int*   u_idx      = (const int*)  d_in[3];
    const float* v_mask     = (const float*)d_in[4];
    const float* fin_ln_b   = (const float*)d_in[5];
    const float* fin_ln_g   = (const float*)d_in[6];
    const float* inp_b      = (const float*)d_in[7];
    const float* inp_w      = (const float*)d_in[8];
    const float* inp_ln_b   = (const float*)d_in[9];
    const float* inp_ln_g   = (const float*)d_in[10];
    // tail params
    const float* uk_b   = (const float*)d_in[75];
    const float* uk_w   = (const float*)d_in[76];
    const float* uq_b   = (const float*)d_in[77];
    const float* uq_w   = (const float*)d_in[78];
    const float* val1_b = (const float*)d_in[79];
    const float* val1_w = (const float*)d_in[80];
    const float* val2_b = (const float*)d_in[81];
    const float* val2_w = (const float*)d_in[82];
    const float* vk_b   = (const float*)d_in[83];
    const float* vk_w   = (const float*)d_in[84];
    const float* vq_b   = (const float*)d_in[85];
    const float* vq_w   = (const float*)d_in[86];

    // workspace carve-out (floats)
    float* ws = (float*)d_ws;
    const size_t SZ = (size_t)MM * DDD;  // 4,194,304
    float* H   = ws;
    float* F   = ws + 1 * SZ;
    float* Qb  = ws + 2 * SZ;
    float* Kb  = ws + 3 * SZ;
    float* Vb  = ws + 4 * SZ;
    float* ATT = ws + 5 * SZ;
    float* MID = ws + 6 * SZ;                       // MM * 1024
    float* G   = MID + (size_t)MM * FF;
    float* QU  = G  + BB * DDD;
    float* T1  = QU + BB * DDD;
    float* QV  = T1 + BB * DDD;
    float* CAT = QV + BB * DDD;                     // BB * 512

    float* u_logits = (float*)d_out;                // 32*512
    float* v_logits = (float*)d_out + MM;           // 32*512
    float* value    = (float*)d_out + 2 * MM;       // 32

    const dim3 gemm_grid_D (DDD / 256, MM / 64);    // N=256 -> (1, 256)
    const dim3 gemm_grid_FF(FF  / 256, MM / 64);    // N=1024 -> (4, 256)
    const dim3 attn_grid(NNN / 16, HHH, BB);

    // input projection + LN + gelu
    input_kernel<<<MM, 256, 0, stream>>>(features, inp_w, inp_b, inp_ln_g, inp_ln_b, H);

    for (int l = 0; l < 4; ++l) {
        const int base = 11 + 16 * l;
        const float* ffn1_b = (const float*)d_in[base + 0];
        const float* ffn1_w = (const float*)d_in[base + 1];
        const float* ffn2_b = (const float*)d_in[base + 2];
        const float* ffn2_w = (const float*)d_in[base + 3];
        const float* k_b    = (const float*)d_in[base + 4];
        const float* k_w    = (const float*)d_in[base + 5];
        const float* ln1_b  = (const float*)d_in[base + 6];
        const float* ln1_g  = (const float*)d_in[base + 7];
        const float* ln2_b  = (const float*)d_in[base + 8];
        const float* ln2_g  = (const float*)d_in[base + 9];
        const float* o_b    = (const float*)d_in[base + 10];
        const float* o_w    = (const float*)d_in[base + 11];
        const float* q_b    = (const float*)d_in[base + 12];
        const float* q_w    = (const float*)d_in[base + 13];
        const float* v_b    = (const float*)d_in[base + 14];
        const float* v_w    = (const float*)d_in[base + 15];

        ln_kernel<<<MM, 256, 0, stream>>>(H, ln1_g, ln1_b, F);
        gemm_wmma<0, false><<<gemm_grid_D, 256, 0, stream>>>(F, q_w, q_b, nullptr, Qb, MM, DDD, DDD);
        gemm_wmma<0, false><<<gemm_grid_D, 256, 0, stream>>>(F, k_w, k_b, nullptr, Kb, MM, DDD, DDD);
        gemm_wmma<0, false><<<gemm_grid_D, 256, 0, stream>>>(F, v_w, v_b, nullptr, Vb, MM, DDD, DDD);
        attn_wmma<<<attn_grid, 32, 0, stream>>>(Qb, Kb, Vb, adj, ATT);
        gemm_wmma<0, true ><<<gemm_grid_D, 256, 0, stream>>>(ATT, o_w, o_b, H, H, MM, DDD, DDD);
        ln_kernel<<<MM, 256, 0, stream>>>(H, ln2_g, ln2_b, F);
        gemm_wmma<1, false><<<gemm_grid_FF, 256, 0, stream>>>(F, ffn1_w, ffn1_b, nullptr, MID, MM, DDD, FF);
        gemm_wmma<0, true ><<<gemm_grid_D, 256, 0, stream>>>(MID, ffn2_w, ffn2_b, H, H, MM, FF, DDD);
    }

    // final LN
    ln_kernel<<<MM, 256, 0, stream>>>(H, fin_ln_g, fin_ln_b, F);
    // pooled graph embedding
    pool_kernel<<<BB, 256, 0, stream>>>(F, valid_mask, G);

    // u logits
    small_linear<<<BB, 256, 0, stream>>>(G, uq_w, uq_b, QU, DDD, DDD, 0);
    gemm_wmma<0, false><<<gemm_grid_D, 256, 0, stream>>>(F, uk_w, uk_b, nullptr, Qb, MM, DDD, DDD);
    logits_kernel<<<MM / 256, 256, 0, stream>>>(QU, Qb, valid_mask, u_logits);

    // value head
    small_linear<<<BB, 256, 0, stream>>>(G, val1_w, val1_b, T1, DDD, DDD, 1);
    val2_kernel<<<1, 32, 0, stream>>>(T1, val2_w, val2_b, value);

    // v logits
    cat_kernel<<<BB, 256, 0, stream>>>(G, F, u_idx, CAT);
    small_linear<<<BB, 256, 0, stream>>>(CAT, vq_w, vq_b, QV, 2 * DDD, DDD, 0);
    gemm_wmma<0, false><<<gemm_grid_D, 256, 0, stream>>>(F, vk_w, vk_b, nullptr, Qb, MM, DDD, DDD);
    logits_kernel<<<MM / 256, 256, 0, stream>>>(QV, Qb, v_mask, v_logits);
}
